// Attention_43447889166824
// MI455X (gfx1250) — matrix operand order repaired and checked by reading, not compiled, not run on hardware
//
#include <hip/hip_runtime.h>
#include <hip/hip_bf16.h>
#include <math.h>

typedef __bf16 bf16;
typedef __attribute__((ext_vector_type(16))) __bf16 v16bf;
typedef __attribute__((ext_vector_type(8)))  float  v8f;

// Problem constants (fixed by the reference): B=2, L=2048, D=1024, H=16, Dh=64
constexpr int kB  = 2;
constexpr int kL  = 2048;   // power of two -> shifts
constexpr int kD  = 1024;
constexpr int kH  = 16;
constexpr int kDh = 64;
constexpr int kM  = kB * kL;  // 4096

static __device__ __forceinline__ v8f vzero8() {
  v8f z = {0.f,0.f,0.f,0.f,0.f,0.f,0.f,0.f};
  return z;
}

static __device__ __forceinline__ v8f wmma_bf16(v16bf a, v16bf b, v8f c) {
  // D = A(16x32) * B(32x16) + C, fp32 accumulate
  return __builtin_amdgcn_wmma_f32_16x16x32_bf16(false, a, false, b, (short)0, c, false, false);
}

// ---------------------------------------------------------------------------
// Fragment loaders per CDNA5 ISA 7.12.2 layouts (wave32).
// A fragment: 16x32 bf16 tile from row-major source (leading dim ld, elements).
// Lane m = lane&15 (both halves), half = lane>>4.
// VGPR j (pair of bf16): k = ((j>>2)<<4) + (half<<3) + ((j&3)<<1)
// ---------------------------------------------------------------------------
static __device__ __forceinline__ v16bf load_frag_a(const bf16* base, int ld,
                                                    int m0, int k0, int lane) {
  int m    = m0 + (lane & 15);
  int half = lane >> 4;
  const unsigned* rp = (const unsigned*)(base + (size_t)m * ld + k0);
  union { v16bf v; unsigned u[8]; } f;
#pragma unroll
  for (int j = 0; j < 8; ++j) {
    int kh = ((j >> 2) << 3) + (half << 2) + (j & 3);  // k/2
    f.u[j] = rp[kh];
  }
  return f.v;
}

// B fragment: B[k][n] = Brow[n][k] with Brow row-major N x K (leading dim ld).
// Lane n = lane&15; same k order as A: VGPR j: k = ((j>>2)<<4) + (half<<3) + ((j&3)<<1)
static __device__ __forceinline__ v16bf load_frag_b(const bf16* Brow, int ld,
                                                    int n0, int k0, int lane) {
  int n    = n0 + (lane & 15);
  int half = lane >> 4;
  const unsigned* rp = (const unsigned*)(Brow + (size_t)n * ld + k0);
  union { v16bf v; unsigned u[8]; } f;
#pragma unroll
  for (int j = 0; j < 8; ++j) f.u[j] = rp[((j >> 2) << 3) + (half << 2) + (j & 3)];
  return f.v;
}

// ---------------------------------------------------------------------------
// fp32 -> bf16 conversion
// ---------------------------------------------------------------------------
__global__ void cvt_f32_bf16_kernel(const float* __restrict__ src,
                                    bf16* __restrict__ dst, size_t n) {
  size_t i = (size_t)blockIdx.x * blockDim.x + threadIdx.x;
  if (i < n) dst[i] = (bf16)src[i];
}

// ---------------------------------------------------------------------------
// Generic C = A @ B^T GEMM over M=4096, N=K=1024 (all constexpr).
// A: M x K bf16 row-major.  Bw: N x K bf16 row-major.
// Each wave computes a 64x64 tile (4x4 C fragments). Block = 4 waves => 128x128.
// MODE 0: bf16 out, head-major [b][h][l][dh]            (q, k)
// MODE 1: bf16 out, transposed head-major [b][h][dh][l] (v)
// MODE 2: fp32 out, row-major [m][n] + bias[n]          (final projection)
// ---------------------------------------------------------------------------
template <int MODE>
__global__ __launch_bounds__(128) void gemm_nt_kernel(
    const bf16* __restrict__ A, const bf16* __restrict__ Bw,
    bf16* __restrict__ outb, float* __restrict__ outf,
    const float* __restrict__ bias) {
  constexpr int K = kD, N = kD;
  int lane = threadIdx.x & 31;
  int wave = threadIdx.x >> 5;
  int m0 = blockIdx.x * 128 + (wave & 1) * 64;
  int n0 = blockIdx.y * 128 + (wave >> 1) * 64;

  v8f c[4][4];
#pragma unroll
  for (int i = 0; i < 4; ++i)
#pragma unroll
    for (int j = 0; j < 4; ++j) c[i][j] = vzero8();

  for (int k0 = 0; k0 < K; k0 += 32) {
    v16bf af[4], bf_[4];
#pragma unroll
    for (int i = 0; i < 4; ++i) af[i]  = load_frag_a(A,  K, m0 + 16 * i, k0, lane);
#pragma unroll
    for (int j = 0; j < 4; ++j) bf_[j] = load_frag_b(Bw, K, n0 + 16 * j, k0, lane);
#pragma unroll
    for (int i = 0; i < 4; ++i)
#pragma unroll
      for (int j = 0; j < 4; ++j) c[i][j] = wmma_bf16(af[i], bf_[j], c[i][j]);
  }

  int half = lane >> 4;
  int nl   = lane & 15;
#pragma unroll
  for (int i = 0; i < 4; ++i) {
#pragma unroll
    for (int j = 0; j < 4; ++j) {
      int n = n0 + 16 * j + nl;
#pragma unroll
      for (int r = 0; r < 8; ++r) {
        int m = m0 + 16 * i + r + 8 * half;
        float v = c[i][j][r];
        if (MODE == 2) {
          outf[(size_t)m * N + n] = v + bias[n];
        } else {
          int btok = m >> 11;          // m / L   (L = 2048)
          int l    = m & (kL - 1);     // m % L
          int h    = n >> 6;           // n / Dh  (Dh = 64)
          int dh   = n & (kDh - 1);    // n % Dh
          size_t idx;
          if (MODE == 0)
            idx = ((((size_t)btok * kH + h) * kL + l) << 6) + dh;
          else
            idx = (((size_t)btok * kH + h) * kDh + dh) * (size_t)kL + l;
          outb[idx] = (bf16)v;
        }
      }
    }
  }
}

// ---------------------------------------------------------------------------
// RoPE on head-major bf16 tensor [bh][l][dh]. One thread per (bh, l, pair).
// ---------------------------------------------------------------------------
__global__ void rope_kernel(bf16* __restrict__ t, const float* __restrict__ tre,
                            const float* __restrict__ tim, size_t total) {
  size_t idx = (size_t)blockIdx.x * blockDim.x + threadIdx.x;
  if (idx >= total) return;
  int p = idx & 31;              // pair index within Dh/2 = 32
  size_t row = idx >> 5;         // bh*L + l
  int l = (int)(row & (kL - 1)); // row % L
  bf16* base = t + (row << 6) + (p << 1);
  float xr = (float)base[0], xi = (float)base[1];
  float tr = tre[l * 32 + p], ti = tim[l * 32 + p];
  base[0] = (bf16)(xr * tr - xi * ti);
  base[1] = (bf16)(xr * ti + xi * tr);
}

// ---------------------------------------------------------------------------
// Flash attention.  Grid: (B*H, L/64), block = 128 threads = 4 waves.
// Each wave owns 16 query rows, streams keys in tiles of 64 (wide tile =>
// one cross-lane butterfly per row per 64 keys instead of two).
// qb/kb: [bh][l][dh] bf16 (post-RoPE).  vtb: [bh][dh][l] bf16.
// ob: [b][l][h*64+dh] bf16 row-major (A operand of the final GEMM).
// ---------------------------------------------------------------------------
__global__ __launch_bounds__(128) void flash_attn_kernel(
    const bf16* __restrict__ qb, const bf16* __restrict__ kb,
    const bf16* __restrict__ vtb, const float* __restrict__ mask,
    bf16* __restrict__ ob) {
  int lane = threadIdx.x & 31;
  int wave = threadIdx.x >> 5;
  int bh = blockIdx.x;
  int b = bh >> 4, h = bh & (kH - 1);
  int q0 = blockIdx.y * 64 + wave * 16;

  const bf16*  Q    = qb  + (size_t)bh * kL * kDh;
  const bf16*  Km   = kb  + (size_t)bh * kL * kDh;
  const bf16*  Vt   = vtb + (size_t)bh * kDh * kL;
  const float* Mrow = mask + (size_t)b * kL * kL;

  __shared__ __align__(16) bf16 lds_p[4][16 * 64];
  bf16* pbuf = lds_p[wave];

  v16bf aq[2];
  aq[0] = load_frag_a(Q, kDh, q0, 0, lane);
  aq[1] = load_frag_a(Q, kDh, q0, 32, lane);

  v8f of[4];
#pragma unroll
  for (int j = 0; j < 4; ++j) of[j] = vzero8();
  float mrun[8], lrun[8];
#pragma unroll
  for (int r = 0; r < 8; ++r) { mrun[r] = -INFINITY; lrun[r] = 0.f; }

  int half = lane >> 4;
  int nl   = lane & 15;
  const float scale = 0.125f;  // 1/sqrt(64)

  for (int kt = 0; kt < kL; kt += 64) {
    // S tile 16x64 = four 16x16 C fragments
    v8f s[4];
#pragma unroll
    for (int nt = 0; nt < 4; ++nt) {
      s[nt] = vzero8();
#pragma unroll
      for (int kk = 0; kk < 2; ++kk)
        s[nt] = wmma_bf16(aq[kk],
                          load_frag_b(Km, kDh, kt + nt * 16, kk * 32, lane),
                          s[nt]);
    }

    // scale + mask
#pragma unroll
    for (int nt = 0; nt < 4; ++nt) {
      int lk = kt + nt * 16 + nl;
#pragma unroll
      for (int r = 0; r < 8; ++r) {
        int lq = q0 + r + 8 * half;
        s[nt][r] = s[nt][r] * scale + Mrow[((size_t)lq << 11) + lk];
      }
    }

    // online softmax per row: combine 4 fragments in registers first, then a
    // single 4-step butterfly across the 16 lanes holding the row.
    float e[4][8];
#pragma unroll
    for (int r = 0; r < 8; ++r) {
      float v = fmaxf(fmaxf(s[0][r], s[1][r]), fmaxf(s[2][r], s[3][r]));
#pragma unroll
      for (int off = 1; off < 16; off <<= 1) v = fmaxf(v, __shfl_xor(v, off, 32));
      float mnew  = fmaxf(mrun[r], v);
      float alpha = __expf(mrun[r] - mnew);
      float rs = 0.f;
#pragma unroll
      for (int nt = 0; nt < 4; ++nt) {
        float a = __expf(s[nt][r] - mnew);
        e[nt][r] = a;
        rs += a;
      }
#pragma unroll
      for (int off = 1; off < 16; off <<= 1) rs += __shfl_xor(rs, off, 32);
      lrun[r] = lrun[r] * alpha + rs;
      mrun[r] = mnew;
#pragma unroll
      for (int j = 0; j < 4; ++j) of[j][r] *= alpha;
    }

    // P (16x64) -> LDS in row-major, reload as two A fragments
#pragma unroll
    for (int r = 0; r < 8; ++r) {
      int m = r + 8 * half;
#pragma unroll
      for (int nt = 0; nt < 4; ++nt)
        pbuf[m * 64 + nt * 16 + nl] = (bf16)e[nt][r];
    }
    v16bf ap[2];
    ap[0] = load_frag_a(pbuf, 64, 0, 0, lane);
    ap[1] = load_frag_a(pbuf, 64, 0, 32, lane);

    // O += P @ V   (B operand from transposed V: contiguous along keys)
#pragma unroll
    for (int j = 0; j < 4; ++j)
#pragma unroll
      for (int kk = 0; kk < 2; ++kk)
        of[j] = wmma_bf16(ap[kk],
                          load_frag_b(Vt, kL, j * 16, kt + kk * 32, lane),
                          of[j]);
  }

  // normalize and store O (16x64) into ob row-major [B*L][1024]
#pragma unroll
  for (int j = 0; j < 4; ++j) {
    int n = h * 64 + j * 16 + nl;
#pragma unroll
    for (int r = 0; r < 8; ++r) {
      int lq = q0 + r + 8 * half;
      size_t m = ((size_t)b << 11) + lq;
      ob[m * kD + n] = (bf16)(of[j][r] / lrun[r]);
    }
  }
}

// ---------------------------------------------------------------------------
// Launcher
// ---------------------------------------------------------------------------
extern "C" void kernel_launch(void* const* d_in, const int* in_sizes, int n_in,
                              void* d_out, int out_size, void* d_ws, size_t ws_size,
                              hipStream_t stream) {
  (void)in_sizes; (void)n_in; (void)out_size;

  const float* x    = (const float*)d_in[0];
  const float* mask = (const float*)d_in[1];
  const float* tre  = (const float*)d_in[2];
  const float* tim  = (const float*)d_in[3];
  const float* Wq   = (const float*)d_in[4];
  const float* Wk   = (const float*)d_in[5];
  const float* Wv   = (const float*)d_in[6];
  const float* Wo   = (const float*)d_in[7];
  const float* bo   = (const float*)d_in[8];
  float* out        = (float*)d_out;

  const size_t sMD = (size_t)kM * kD;   // 4M elements
  const size_t sDD = (size_t)kD * kD;   // 1M elements
  const size_t need = (5 * sMD + 4 * sDD) * sizeof(bf16);  // 48 MB
  if (ws_size < need) return;

  bf16* ws  = (bf16*)d_ws;
  size_t off = 0;
  bf16* xb   = ws + off; off += sMD;
  bf16* Wqb  = ws + off; off += sDD;
  bf16* Wkb  = ws + off; off += sDD;
  bf16* Wvb  = ws + off; off += sDD;
  bf16* Wob  = ws + off; off += sDD;
  bf16* qb   = ws + off; off += sMD;
  bf16* kbuf = ws + off; off += sMD;
  bf16* vtb  = ws + off; off += sMD;
  bf16* obuf = ws + off; off += sMD;

  // 1) fp32 -> bf16
  cvt_f32_bf16_kernel<<<(int)((sMD + 255) / 256), 256, 0, stream>>>(x,  xb,  sMD);
  cvt_f32_bf16_kernel<<<(int)((sDD + 255) / 256), 256, 0, stream>>>(Wq, Wqb, sDD);
  cvt_f32_bf16_kernel<<<(int)((sDD + 255) / 256), 256, 0, stream>>>(Wk, Wkb, sDD);
  cvt_f32_bf16_kernel<<<(int)((sDD + 255) / 256), 256, 0, stream>>>(Wv, Wvb, sDD);
  cvt_f32_bf16_kernel<<<(int)((sDD + 255) / 256), 256, 0, stream>>>(Wo, Wob, sDD);

  // 2) projections
  dim3 ggrid(kM / 128, kD / 128);  // 32 x 8
  gemm_nt_kernel<0><<<ggrid, 128, 0, stream>>>(xb, Wqb, qb,   nullptr, nullptr);
  gemm_nt_kernel<0><<<ggrid, 128, 0, stream>>>(xb, Wkb, kbuf, nullptr, nullptr);
  gemm_nt_kernel<1><<<ggrid, 128, 0, stream>>>(xb, Wvb, vtb,  nullptr, nullptr);

  // 3) RoPE on q and k
  size_t rope_total = (size_t)kB * kH * kL * 32;
  rope_kernel<<<(int)((rope_total + 255) / 256), 256, 0, stream>>>(qb,   tre, tim, rope_total);
  rope_kernel<<<(int)((rope_total + 255) / 256), 256, 0, stream>>>(kbuf, tre, tim, rope_total);

  // 4) flash attention
  dim3 fgrid(kB * kH, kL / 64);  // 32 x 32
  flash_attn_kernel<<<fgrid, 128, 0, stream>>>(qb, kbuf, vtb, mask, obuf);

  // 5) output projection + bias (fp32 out)
  gemm_nt_kernel<2><<<ggrid, 128, 0, stream>>>(obuf, Wob, nullptr, out, bo);
}
